// AttentionGNN_3539053052128
// MI455X (gfx1250) — compile-verified, matrix operand order checked by reading, not run on hardware
//
#include <hip/hip_runtime.h>
#include <hip/hip_bf16.h>

typedef __attribute__((ext_vector_type(16))) _Float16 v16h;
typedef __attribute__((ext_vector_type(8)))  _Float16 v8h;
typedef __attribute__((ext_vector_type(8)))  float    v8f;

// ---------------------------------------------------------------------------
// Wave-per-(16 x 16*NT)-strip WMMA GEMM:  C[M,N] = act(A[M,K]) @ Bt[N,K]^T (+bias)
// Requirements (guaranteed by host-side packing):
//   M % 16 == 0, N % (16*NT) == 0, K % 32 == 0
//   A: f32 row-major [M, lda], lda % 4 == 0 (16B-aligned rows)
//   Bt: f16 row-major [N, K]  (logical B[K,N] transposed, K zero-padded)
// LRELU: leaky_relu(0.01) == max(x, 0.01x) fused on A load (compile-time).
// NT: n-tiles per wave; A loaded/converted once per slab, NT wmmas issued.
// ---------------------------------------------------------------------------
template <bool LRELU, int NT>
__global__ void gemm_wmma(const float* __restrict__ A, const _Float16* __restrict__ Bt,
                          const float* __restrict__ bias, float* __restrict__ C,
                          int M, int Ncols, int K, int lda, int ldc,
                          long strideA, long strideB, long strideC)
{
    int batch = blockIdx.y;
    A  += (long)batch * strideA;
    Bt += (long)batch * strideB;
    C  += (long)batch * strideC;

    int wave = blockIdx.x * (blockDim.x >> 5) + (threadIdx.x >> 5);
    int Mt = M >> 4;
    int Ng = (Ncols >> 4) / NT;                 // groups of NT n-tiles
    if (wave >= Mt * Ng) return;                // wave-uniform: EXEC all-1s for WMMA
    int tm = wave / Ng, tg = wave % Ng;
    int m0 = tm * 16, n0 = tg * (16 * NT);

    int lane  = threadIdx.x & 31;
    int bl    = lane & 15;
    int khalf = (lane >> 4) ? 8 : 0;            // lanes 16..31 own K 8..15 / 24..31

    const float* Ap = A + (m0 + bl) * lda + khalf;
    const _Float16* Bp[NT];
#pragma unroll
    for (int j = 0; j < NT; ++j)
        Bp[j] = Bt + (n0 + j * 16 + bl) * K + khalf;

    // uniform speculative prefetch of the second K-slab
    if (K > 32) {
        __builtin_prefetch(Ap + 32, 0, 0);
#pragma unroll
        for (int j = 0; j < NT; ++j) __builtin_prefetch(Bp[j] + 32, 0, 0);
    }

    v8f acc[NT];
#pragma unroll
    for (int j = 0; j < NT; ++j) acc[j] = (v8f){};

    for (int k0 = 0; k0 < K; k0 += 32) {
        // A: 16 f32 per lane, groups [khalf..+7] and [khalf+16..+23]
        float4 a0 = *(const float4*)(Ap + k0);
        float4 a1 = *(const float4*)(Ap + k0 + 4);
        float4 a2 = *(const float4*)(Ap + k0 + 16);
        float4 a3 = *(const float4*)(Ap + k0 + 20);
        float af[16] = { a0.x, a0.y, a0.z, a0.w, a1.x, a1.y, a1.z, a1.w,
                         a2.x, a2.y, a2.z, a2.w, a3.x, a3.y, a3.z, a3.w };
        v16h av;
#pragma unroll
        for (int e = 0; e < 16; ++e) {
            float va = af[e];
            if (LRELU) va = fmaxf(va, 0.01f * va);   // leaky-relu, no cmp/cndmask
            av[e] = (_Float16)va;
        }
#pragma unroll
        for (int j = 0; j < NT; ++j) {
            v8h b0 = *(const v8h*)(Bp[j] + k0);
            v8h b1 = *(const v8h*)(Bp[j] + k0 + 16);
            v16h bv;
#pragma unroll
            for (int e = 0; e < 8; ++e) { bv[e] = b0[e]; bv[e + 8] = b1[e]; }
            acc[j] = __builtin_amdgcn_wmma_f32_16x16x32_f16(false, av, false, bv,
                                                            (short)0, acc[j], false, false);
        }
    }

    // 32-bit C/D layout: lane<16 rows m0..m0+7, lane>=16 rows m0+8..m0+15
    int rbase = m0 + ((lane >> 4) ? 8 : 0);
#pragma unroll
    for (int j = 0; j < NT; ++j) {
        int bcol = n0 + j * 16 + bl;
        float bvv = bias ? bias[bcol] : 0.f;
        int base = rbase * ldc + bcol;               // fits in 32 bits for all our C
#pragma unroll
        for (int r = 0; r < 8; ++r)
            C[base + r * ldc] = acc[j][r] + bvv;
    }
}

// ---------------------------------------------------------------------------
// Packing / elementwise helpers
// ---------------------------------------------------------------------------
__global__ void pad_rows(const float* __restrict__ src, float* __restrict__ dst,
                         int rows, int incols, int outcols)
{
    int idx = blockIdx.x * blockDim.x + threadIdx.x;
    if (idx >= rows * outcols) return;
    int r = idx / outcols, c = idx % outcols;
    dst[idx] = (c < incols) ? src[(long)r * incols + c] : 0.f;
}

// Bt[n, k] = (k < Kin) ? W[k, n] : 0, f16.  W is [Kin, Nc] row-major.
__global__ void pack_weight_f16(const float* __restrict__ W, _Float16* __restrict__ Bt,
                                int Kin, int Kpad, int Nc)
{
    int idx = blockIdx.x * blockDim.x + threadIdx.x;
    if (idx >= Nc * Kpad) return;
    int n = idx / Kpad, k = idx % Kpad;
    Bt[idx] = (_Float16)((k < Kin) ? W[(long)k * Nc + n] : 0.f);
}

// Generated-weight pack (transposed, f16): Bt[c, i], c in [0,1088), i in [0,64)
__global__ void repack_efW_f16(const float* __restrict__ efW, const float* __restrict__ efb,
                               _Float16* __restrict__ Bt)
{
    int idx = blockIdx.x * blockDim.x + threadIdx.x;  // 1088*64
    if (idx >= 1088 * 64) return;
    int c = idx >> 6, i = idx & 63;
    float v;
    if (c < 1024) { int k = c >> 6, o = c & 63; v = efW[k * 4096 + i * 64 + o]; }
    else          { v = efb[i * 64 + (c - 1024)]; }
    Bt[idx] = (_Float16)v;
}

__global__ void cvt_f32_f16(const float* __restrict__ s, _Float16* __restrict__ d, int n)
{
    int i = blockIdx.x * blockDim.x + threadIdx.x;
    if (i < n) d[i] = (_Float16)s[i];
}

// LayerNorm over rows of 64, optional residual add. One thread per row.
__global__ void ln_rows(const float* __restrict__ x, const float* __restrict__ g,
                        const float* __restrict__ b, const float* __restrict__ residual,
                        float* __restrict__ y, int rows)
{
    int r = blockIdx.x * blockDim.x + threadIdx.x;
    if (r >= rows) return;
    const float* xr = x + (long)r * 64;
    float m = 0.f;
    for (int i = 0; i < 64; ++i) m += xr[i];
    m *= (1.f / 64.f);
    float v = 0.f;
    for (int i = 0; i < 64; ++i) { float d = xr[i] - m; v += d * d; }
    v *= (1.f / 64.f);
    float inv = rsqrtf(v + 1e-5f);
    float* yr = y + (long)r * 64;
    for (int i = 0; i < 64; ++i) {
        float o = (xr[i] - m) * inv * g[i] + b[i];
        if (residual) o += residual[(long)r * 64 + i];
        yr[i] = o;
    }
}

__global__ void fill_zero(float* __restrict__ p, int n)
{
    int i = blockIdx.x * blockDim.x + threadIdx.x;
    if (i < n) p[i] = 0.f;
}

__global__ void count_deg(const int* __restrict__ dst, float* __restrict__ deg, int E)
{
    int e = blockIdx.x * blockDim.x + threadIdx.x;
    if (e < E) atomicAdd(&deg[dst[e]], 1.f);
}

// Per-edge: msg[o] = Tb[src,o] + sum_k eh[e,k]*T[src,k*64+o]; atomic scatter to agg[dst].
__global__ void edge_msg(const float* __restrict__ eh, const int* __restrict__ src,
                         const int* __restrict__ dst, const float* __restrict__ T,
                         float* __restrict__ agg, int E)
{
    int e = blockIdx.x * (blockDim.x >> 6) + (threadIdx.x >> 6);
    int o = threadIdx.x & 63;
    if (e >= E) return;
    int s = src[e], d = dst[e];
    const float* Ts = T + (long)s * 1088;
    const float* ehe = eh + (long)e * 16;
    float m = Ts[1024 + o];
#pragma unroll
    for (int k = 0; k < 16; ++k) m = fmaf(ehe[k], Ts[k * 64 + o], m);
    atomicAdd(&agg[(long)d * 64 + o], m);
}

// agg = lrelu(agg / max(deg,1) + convb)
__global__ void finish_agg(float* __restrict__ agg, const float* __restrict__ deg,
                           const float* __restrict__ convb, int n64)
{
    int i = blockIdx.x * blockDim.x + threadIdx.x;
    if (i >= n64) return;
    int n = i >> 6, o = i & 63;
    float dv = fmaxf(deg[n], 1.f);
    float v = agg[i] / dv + convb[o];
    agg[i] = fmaxf(v, 0.01f * v);
}

// out[b] = mean(op[b, :, :]) over 64*256 entries
__global__ void reduce_op(const float* __restrict__ op, float* __restrict__ out)
{
    __shared__ float s[256];
    int b = blockIdx.x;
    const float* p = op + (long)b * 16384;
    float acc = 0.f;
    for (int i = threadIdx.x; i < 16384; i += 256) acc += p[i];
    s[threadIdx.x] = acc;
    __syncthreads();
    for (int st = 128; st > 0; st >>= 1) {
        if (threadIdx.x < st) s[threadIdx.x] += s[threadIdx.x + st];
        __syncthreads();
    }
    if (threadIdx.x == 0) out[b] = s[0] * (1.f / 16384.f);
}

// ---------------------------------------------------------------------------
// Host side
// ---------------------------------------------------------------------------
static const int B_ = 16, L_ = 64, R_ = 256;
static const int NLIG = B_ * L_, NREC = B_ * R_;
static const int ELIG = NLIG * 8, EREC = NREC * 8;

// workspace offsets (floats), sized for the larger (rec) side and reused
static const long OFF_EH    = 0;                       // E*16      (<= 524288)
static const long OFF_H     = OFF_EH    + 524288;      // N*64      (<= 262144)
static const long OFF_TMP   = OFF_H     + 262144;
static const long OFF_TMP2  = OFF_TMP   + 262144;
static const long OFF_PREV0 = OFF_TMP2  + 262144;
static const long OFF_T     = OFF_PREV0 + 262144;      // N*1088    (<= 4456448)
static const long OFF_DEG   = OFF_T     + 4456448;     // N         (<= 4096)
static const long OFF_LIGH  = OFF_DEG   + 4096;        // 1024*64 = 65536
static const long OFF_XPAD  = OFF_LIGH  + 65536;       // N*64 padded node feats
static const long OFF_EPAD  = OFF_XPAD  + 262144;      // E*32 padded edge feats (<=1048576)
static const long OFF_HALF  = OFF_EPAD  + 1048576;     // f16 region below (in halves)
// f16 region offsets (halves)
static const long HOFF_WP    = 0;                      // 1088*64 = 69632
static const long HOFF_WN    = HOFF_WP + 69632;        // 64*64
static const long HOFF_WE    = HOFF_WN + 4096;         // 16*32
static const long HOFF_WH    = HOFF_WE + 512;          // 64*64
static const long HOFF_WO    = HOFF_WH + 4096;         // 64*64
static const long HOFF_RECHF = HOFF_WO + 4096;         // 4096*64 = 262144

static inline void launch_gemm(hipStream_t s, const float* A, const _Float16* Bt,
                               const float* bias, float* C,
                               int M, int N, int K, int lda, int ldc,
                               long sA, long sB, long sC, int batches, bool lrelu)
{
    int Nt = N >> 4;
    if ((Nt & 3) == 0) {
        int waves = (M >> 4) * (Nt >> 2);
        dim3 grid((waves + 3) / 4, batches);       // 128 threads = 4 wave32/block
        if (lrelu) gemm_wmma<true, 4><<<grid, 128, 0, s>>>(A, Bt, bias, C, M, N, K, lda, ldc, sA, sB, sC);
        else       gemm_wmma<false, 4><<<grid, 128, 0, s>>>(A, Bt, bias, C, M, N, K, lda, ldc, sA, sB, sC);
    } else {
        int waves = (M >> 4) * Nt;
        dim3 grid((waves + 3) / 4, batches);
        if (lrelu) gemm_wmma<true, 1><<<grid, 128, 0, s>>>(A, Bt, bias, C, M, N, K, lda, ldc, sA, sB, sC);
        else       gemm_wmma<false, 1><<<grid, 128, 0, s>>>(A, Bt, bias, C, M, N, K, lda, ldc, sA, sB, sC);
    }
}

static void run_side(hipStream_t s, float* ws, int N, int E, int NIN,
                     const float* x, const float* e, const int* src, const int* dst,
                     const float* emb_nW, const float* emb_nb,
                     const float* emb_eW, const float* emb_eb,
                     const float* hidW, const float* hidb,
                     const float* ln0_g, const float* ln0_b,
                     const float* efW, const float* efb, const float* convb,
                     const float* outW, const float* outb,
                     const float* lng, const float* lnb)
{
    float* eh    = ws + OFF_EH;
    float* h     = ws + OFF_H;
    float* tmp   = ws + OFF_TMP;
    float* tmp2  = ws + OFF_TMP2;
    float* prev0 = ws + OFF_PREV0;
    float* T     = ws + OFF_T;
    float* deg   = ws + OFF_DEG;
    float* xpad  = ws + OFF_XPAD;
    float* epad  = ws + OFF_EPAD;
    _Float16* hb = (_Float16*)(ws + OFF_HALF);
    _Float16* WpT = hb + HOFF_WP;
    _Float16* wN  = hb + HOFF_WN;
    _Float16* wE  = hb + HOFF_WE;
    _Float16* wH  = hb + HOFF_WH;
    _Float16* wO  = hb + HOFF_WO;

    // pad inputs to WMMA-friendly K, pre-pack weights transposed+f16
    pad_rows<<<(N * 64 + 255) / 256, 256, 0, s>>>(x, xpad, N, NIN, 64);
    pad_rows<<<(E * 32 + 255) / 256, 256, 0, s>>>(e, epad, E, 16, 32);
    pack_weight_f16<<<(64 * 64 + 255) / 256, 256, 0, s>>>(emb_nW, wN, NIN, 64, 64);
    pack_weight_f16<<<(16 * 32 + 255) / 256, 256, 0, s>>>(emb_eW, wE, 16, 32, 16);
    pack_weight_f16<<<(64 * 64 + 255) / 256, 256, 0, s>>>(hidW, wH, 64, 64, 64);

    // edge embedding: eh = e @ emb_eW + emb_eb   [E,16]
    launch_gemm(s, epad, wE, emb_eb, eh, E, 16, 32, 32, 16, 0, 0, 0, 1, false);
    // node embedding chain
    launch_gemm(s, xpad, wN, emb_nb, tmp, N, 64, 64, 64, 64, 0, 0, 0, 1, false);
    launch_gemm(s, tmp, wH, hidb, tmp2, N, 64, 64, 64, 64, 0, 0, 0, 1, true);
    ln_rows<<<(N + 63) / 64, 64, 0, s>>>(tmp2, ln0_g, ln0_b, nullptr, h, N);

    fill_zero<<<(N + 255) / 256, 256, 0, s>>>(deg, N);
    count_deg<<<(E + 255) / 256, 256, 0, s>>>(dst, deg, E);

    for (int l = 0; l < 3; ++l) {
        // per-node generated-weight precompute (DEG x FLOP cut vs per-edge We):
        // T[n, k*64+o] = sum_i lrelu(h)[n,i] * efW[l][k, i*64+o]; cols 1024.. carry efb
        repack_efW_f16<<<(1088 * 64 + 255) / 256, 256, 0, s>>>(efW + (long)l * 16 * 4096,
                                                               efb + (long)l * 4096, WpT);
        launch_gemm(s, h, WpT, nullptr, T, N, 1088, 64, 64, 1088, 0, 0, 0, 1, true);
        fill_zero<<<(N * 64 + 255) / 256, 256, 0, s>>>(tmp, N * 64);
        edge_msg<<<(E + 3) / 4, 256, 0, s>>>(eh, src, dst, T, tmp, E);
        finish_agg<<<(N * 64 + 255) / 256, 256, 0, s>>>(tmp, deg, convb + l * 64, N * 64);
        pack_weight_f16<<<(64 * 64 + 255) / 256, 256, 0, s>>>(outW + (long)l * 64 * 64,
                                                              wO, 64, 64, 64);
        launch_gemm(s, tmp, wO, outb + l * 64, tmp2, N, 64, 64, 64, 64, 0, 0, 0, 1, false);
        ln_rows<<<(N + 63) / 64, 64, 0, s>>>(tmp2, lng + l * 64, lnb + l * 64,
                                             (l == 2) ? prev0 : nullptr, h, N);
        if (l == 0)
            (void)hipMemcpyAsync(prev0, h, (size_t)N * 64 * sizeof(float),
                                 hipMemcpyDeviceToDevice, s);
    }
}

extern "C" void kernel_launch(void* const* d_in, const int* in_sizes, int n_in,
                              void* d_out, int out_size, void* d_ws, size_t ws_size,
                              hipStream_t stream)
{
    (void)in_sizes; (void)n_in; (void)out_size; (void)ws_size;
    const float* lig_x = (const float*)d_in[0];
    const float* rec_x = (const float*)d_in[1];
    const float* lig_e = (const float*)d_in[2];
    const float* rec_e = (const float*)d_in[3];
    const int* lig_src = (const int*)d_in[4];
    const int* lig_dst = (const int*)d_in[5];
    const int* rec_src = (const int*)d_in[6];
    const int* rec_dst = (const int*)d_in[7];
    const float* P[30];
    for (int i = 0; i < 30; ++i) P[i] = (const float*)d_in[8 + i];
    const float** Lp = P;        // lig params [0..14]
    const float** Rp = P + 15;   // rec params [0..14]

    float* ws = (float*)d_ws;
    float* out = (float*)d_out;       // [16]
    float* op  = out + 16;            // [16, 64, 256]

    // ligand tower, stash final h
    run_side(stream, ws, NLIG, ELIG, 32, lig_x, lig_e, lig_src, lig_dst,
             Lp[0], Lp[1], Lp[2], Lp[3], Lp[4], Lp[5], Lp[6], Lp[7],
             Lp[8], Lp[9], Lp[10], Lp[11], Lp[12], Lp[13], Lp[14]);
    (void)hipMemcpyAsync(ws + OFF_LIGH, ws + OFF_H, (size_t)NLIG * 64 * sizeof(float),
                         hipMemcpyDeviceToDevice, stream);

    // receptor tower (reuses workspace); final h stays at OFF_H
    run_side(stream, ws, NREC, EREC, 40, rec_x, rec_e, rec_src, rec_dst,
             Rp[0], Rp[1], Rp[2], Rp[3], Rp[4], Rp[5], Rp[6], Rp[7],
             Rp[8], Rp[9], Rp[10], Rp[11], Rp[12], Rp[13], Rp[14]);

    // scoring: op[b] = lig_h[b] (64x64) @ rec_h[b]^T; rec_h already [R,K] -> f16 Bt
    _Float16* recHf = (_Float16*)(ws + OFF_HALF) + HOFF_RECHF;
    cvt_f32_f16<<<(NREC * 64 + 255) / 256, 256, 0, stream>>>(ws + OFF_H, recHf, NREC * 64);
    launch_gemm(stream, ws + OFF_LIGH, recHf, nullptr, op,
                /*M=*/64, /*N=*/256, /*K=*/64, /*lda=*/64, /*ldc=*/256,
                /*sA=*/64 * 64, /*sB=*/256 * 64, /*sC=*/64 * 256, /*batches=*/16, false);

    // out[b] = mean(op[b])
    reduce_op<<<16, 256, 0, stream>>>(op, out);
}
